// Graph_GCN_75780402970867
// MI455X (gfx1250) — matrix operand order repaired and checked by reading, not compiled, not run on hardware
//
#include <hip/hip_runtime.h>
#include <stdint.h>

// ---------------- problem constants ----------------
#define V      8192
#define BB     32
#define F0     16
#define NCOL   512          // BB * F0 folded column block
#define KPOLY  8
#define CL1F   32
#define POOLW  8
#define FC1FIN 32768        // CL1F * (V/POOLW)
#define FC1F   512
#define NNF1   256
#define NNF2   128
#define OUTF   10
#define BN_EPS 1e-5f

typedef __attribute__((ext_vector_type(16))) __bf16 v16bf;
typedef __attribute__((ext_vector_type(8)))  float  v8f;
typedef __attribute__((ext_vector_type(4)))  unsigned int u32x4;
typedef __attribute__((ext_vector_type(4)))  int          i32x4;
typedef __attribute__((ext_vector_type(8)))  int          i32x8;

static __device__ __forceinline__ uint16_t f32_to_bf16(float f) {
    uint32_t u = __float_as_uint(f);
    uint32_t r = u + 0x7FFFu + ((u >> 16) & 1u);   // round-to-nearest-even
    return (uint16_t)(r >> 16);
}

// ---------------- Bernstein coefficients: a = M(K) @ theta ----------------
__global__ void bern_kernel(const float* __restrict__ theta, float* __restrict__ a) {
    if (threadIdx.x == 0) {
        for (int j = 0; j <= KPOLY; ++j) {
            double s = 0.0;
            for (int k = 0; k <= j; ++k) {
                int n = KPOLY - k, r = j - k;
                double c = 1.0;
                for (int i = 1; i <= r; ++i) c = c * (double)(n - r + i) / (double)i;
                double p2 = 1.0;
                for (int i = 0; i < KPOLY - j; ++i) p2 *= 2.0;
                double sign = ((j - k) & 1) ? -1.0 : 1.0;
                s += (double)theta[k] * c * p2 * sign;
            }
            a[j] = (float)s;
        }
    }
}

// ---------------- fp32 -> bf16 bulk convert (8 elems / thread) ----------------
__global__ void cvt_bf16_kernel(const float* __restrict__ src,
                                uint16_t* __restrict__ dst, long long n8) {
    long long t = (long long)blockIdx.x * blockDim.x + threadIdx.x;
    if (t >= n8) return;
    const float4* s = (const float4*)src + t * 2;
    float4 x0 = s[0], x1 = s[1];
    uint32_t w0 = f32_to_bf16(x0.x) | ((uint32_t)f32_to_bf16(x0.y) << 16);
    uint32_t w1 = f32_to_bf16(x0.z) | ((uint32_t)f32_to_bf16(x0.w) << 16);
    uint32_t w2 = f32_to_bf16(x1.x) | ((uint32_t)f32_to_bf16(x1.y) << 16);
    uint32_t w3 = f32_to_bf16(x1.z) | ((uint32_t)f32_to_bf16(x1.w) << 16);
    ((uint4*)dst)[t] = make_uint4(w0, w1, w2, w3);
}

// ---------------- OutAcc = a0 * x, laid out (V, 512) ----------------
__global__ void init_acc_kernel(const float* __restrict__ x, const float* __restrict__ a,
                                float* __restrict__ OutAcc) {
    int t = blockIdx.x * blockDim.x + threadIdx.x;     // V*NCOL threads exactly
    int v = t / NCOL, c = t % NCOL;
    int b = c / F0, f = c % F0;
    OutAcc[t] = a[0] * x[(size_t)b * V * F0 + (size_t)v * F0 + f];
}

// ---------------- W0^T bf16, laid out (512, V): row c holds w(:,c) over v -----------
__global__ void init_wt_kernel(const float* __restrict__ x, uint16_t* __restrict__ W0t) {
    int t = blockIdx.x * blockDim.x + threadIdx.x;     // NCOL*V threads exactly
    int c = t / V, v = t % V;
    int b = c / F0, f = c % F0;
    W0t[t] = f32_to_bf16(x[(size_t)b * V * F0 + (size_t)v * F0 + f]);
}

// ---------------- TDM helper: 2-D tile (nrows x 32 bf16) -> padded LDS ----------------
#define BM  128
#define BN  128
#define BK  32
#define LDP 40   // padded LDS row stride (elements): 64B row + 16B TDM pad

static __device__ __forceinline__ void tdm_load_2d(
    uint32_t lds_byte_off, const uint16_t* gp,
    uint32_t tensor_d0, uint32_t tensor_d1, uint32_t stride_elems, uint32_t nrows)
{
    uint64_t ga = (uint64_t)(uintptr_t)gp;
    u32x4 g0 = { 1u,                                    // count=1 valid user D#
                 lds_byte_off,                          // lds_addr
                 (uint32_t)ga,                          // global_addr[31:0]
                 ((uint32_t)(ga >> 32) & 0x01FFFFFFu)   // global_addr[56:32]
                   | (2u << 30) };                      // type=2 ("image")
    i32x8 g1;
    g1[0] = (int)((1u << 16)        // data_size = 2 bytes
                | (1u << 20)        // pad_enable
                | (3u << 22)        // pad_interval: 16 DWORDs (=64B, one tile row)
                | (3u << 25));      // pad_amount: 4 DWORDs (=16B) -> LDS stride 80B
    g1[1] = (int)(tensor_d0 << 16);                       // tensor_dim0[15:0]
    g1[2] = (int)((tensor_d0 >> 16) | (tensor_d1 << 16)); // td0[31:16] | td1[15:0]
    g1[3] = (int)((tensor_d1 >> 16) | ((uint32_t)BK << 16)); // td1[31:16] | tile_dim0=32
    g1[4] = (int)nrows;                                   // tile_dim1, tile_dim2=0
    g1[5] = (int)stride_elems;                            // tensor_dim0_stride[31:0]
    g1[6] = 0;                                            // stride[47:32]=0
    g1[7] = 0;
    i32x4 gz = { 0, 0, 0, 0 };                            // 2-D: groups 2/3 unused
#if __has_include(<hip/amd_detail/amd_gfx1250_TDM.h>)
    i32x8 gz8 = { 0, 0, 0, 0, 0, 0, 0, 0 };
    __builtin_amdgcn_tensor_load_to_lds(g0, g1, gz, gz, gz8, 0);
#else
    __builtin_amdgcn_tensor_load_to_lds(g0, g1, gz, gz, 0);
#endif
}

// ---------------- core WMMA GEMM: C = L @ W ; OutAcc += a[j]*C ; Wn^T = bf16(C) -------
// L:  (V,V) bf16 row-major            -> A tiles (128 rows x 32 k)
// Wt: (NCOL,V) bf16 (transposed W)    -> B tiles (128 cols x 32 k), row-contiguous
// Both tiles fetched by the Tensor Data Mover into double-buffered padded LDS,
// overlapped with WMMA on the other buffer.
__global__ __launch_bounds__(256) void gemm_bf16_kernel(
    const uint16_t* __restrict__ Lb,
    const uint16_t* __restrict__ Wct,    // current W^T (NCOL, V)
    uint16_t* __restrict__ Wnt,          // next W^T (NCOL, V) out
    float* __restrict__ OutAcc,          // (V, NCOL) f32, accumulated
    const float* __restrict__ a, int j)
{
    __shared__ __align__(128) uint16_t As[2][BM * LDP];   // [row][k], TDM-padded
    __shared__ __align__(128) uint16_t Bs[2][BN * LDP];   // [col][k], TDM-padded

    const int tid   = threadIdx.x;
    const int lane  = tid & 31;
    const int wave  = tid >> 5;          // 0..7
    const int wm    = wave >> 2;         // 0..1 : 64-row slab
    const int wn    = wave & 3;          // 0..3 : 32-col slab
    const int bm    = blockIdx.x;        // 0..63
    const int bn    = blockIdx.y;        // 0..3
    const int lhalf = lane >> 4;         // 0/1
    const int l15   = lane & 15;

    const v8f vzero = {0.f,0.f,0.f,0.f,0.f,0.f,0.f,0.f};
    v8f acc[4][2];
#pragma unroll
    for (int mt = 0; mt < 4; ++mt)
#pragma unroll
        for (int nt = 0; nt < 2; ++nt) acc[mt][nt] = vzero;

    const uint16_t* Atile = Lb  + (size_t)(bm * BM) * V;   // + k elems per step
    const uint16_t* Btile = Wct + (size_t)(bn * BN) * V;
    const uint32_t  asOff[2] = { (uint32_t)(uintptr_t)&As[0][0], (uint32_t)(uintptr_t)&As[1][0] };
    const uint32_t  bsOff[2] = { (uint32_t)(uintptr_t)&Bs[0][0], (uint32_t)(uintptr_t)&Bs[1][0] };

    // prologue: DMA first tiles into buffer 0 (wave 0 drives the TDM)
    if (wave == 0) {
        tdm_load_2d(asOff[0], Atile, V, V, V, BM);
        tdm_load_2d(bsOff[0], Btile, V, NCOL, V, BN);
    }

    const int NK = V / BK;               // 256 k-steps
    for (int kk = 0; kk < NK; ++kk) {
        const int cur = kk & 1;
        if (wave == 0) {
            if (kk + 1 < NK) {           // prefetch next tiles, then drain current pair
                const int kn = (kk + 1) * BK;
                tdm_load_2d(asOff[cur ^ 1], Atile + kn, V, V, V, BM);
                tdm_load_2d(bsOff[cur ^ 1], Btile + kn, V, NCOL, V, BN);
                __builtin_amdgcn_s_wait_tensorcnt(2);
            } else {
                __builtin_amdgcn_s_wait_tensorcnt(0);
            }
        }
        __syncthreads();                  // publish TDM-written LDS to all waves

        union Frag { uint4 q[2]; v16bf v; };
        Frag af[4], bf[2];
#pragma unroll
        for (int mt = 0; mt < 4; ++mt) {
            int r = wm * 64 + mt * 16 + l15;
            af[mt].q[0] = *(const uint4*)&As[cur][r * LDP + lhalf * 8];       // K{0..7}/{8..15}
            af[mt].q[1] = *(const uint4*)&As[cur][r * LDP + 16 + lhalf * 8];  // K{16..23}/{24..31}
        }
#pragma unroll
        for (int nt = 0; nt < 2; ++nt) {
            int c = wn * 32 + nt * 16 + l15;
            bf[nt].q[0] = *(const uint4*)&Bs[cur][c * LDP + lhalf * 8];       // K{0..7}/{8..15}
            bf[nt].q[1] = *(const uint4*)&Bs[cur][c * LDP + 16 + lhalf * 8];  // K{16..23}/{24..31}
        }
#pragma unroll
        for (int mt = 0; mt < 4; ++mt)
#pragma unroll
            for (int nt = 0; nt < 2; ++nt)
                acc[mt][nt] = __builtin_amdgcn_wmma_f32_16x16x32_bf16(
                    false, af[mt].v, false, bf[nt].v,
                    (short)0, acc[mt][nt], false, false);
        __syncthreads();                  // all reads done before TDM reuses this buffer
    }

    const float aj = a[j];
#pragma unroll
    for (int mt = 0; mt < 4; ++mt) {
#pragma unroll
        for (int nt = 0; nt < 2; ++nt) {
            const int row0 = bm * BM + wm * 64 + mt * 16 + lhalf * 8;
            const int col  = bn * BN + wn * 32 + nt * 16 + l15;
            // f32 accumulation into (V, NCOL)
#pragma unroll
            for (int i = 0; i < 8; ++i)
                OutAcc[(size_t)(row0 + i) * NCOL + col] += aj * acc[mt][nt][i];
            // bf16 next-signal, transposed (NCOL, V): 8 consecutive rows -> one b128 store
            uint32_t p0 = (uint32_t)f32_to_bf16(acc[mt][nt][0]) | ((uint32_t)f32_to_bf16(acc[mt][nt][1]) << 16);
            uint32_t p1 = (uint32_t)f32_to_bf16(acc[mt][nt][2]) | ((uint32_t)f32_to_bf16(acc[mt][nt][3]) << 16);
            uint32_t p2 = (uint32_t)f32_to_bf16(acc[mt][nt][4]) | ((uint32_t)f32_to_bf16(acc[mt][nt][5]) << 16);
            uint32_t p3 = (uint32_t)f32_to_bf16(acc[mt][nt][6]) | ((uint32_t)f32_to_bf16(acc[mt][nt][7]) << 16);
            *(uint4*)&Wnt[(size_t)col * V + row0] = make_uint4(p0, p1, p2, p3);
        }
    }
}

// ---------------- column sum-of-squares over V (for F.normalize dim=1) ----------------
__global__ void zero_kernel(float* __restrict__ p, int n) {
    int t = blockIdx.x * blockDim.x + threadIdx.x;
    if (t < n) p[t] = 0.f;
}

__global__ void colsumsq_kernel(const float* __restrict__ A, float* __restrict__ nrm) {
    int c0 = threadIdx.x, c1 = threadIdx.x + 256;
    int r0 = blockIdx.x * 128;
    float s0 = 0.f, s1 = 0.f;
    for (int r = r0; r < r0 + 128; ++r) {
        float x0 = A[(size_t)r * NCOL + c0];
        float x1 = A[(size_t)r * NCOL + c1];
        s0 += x0 * x0; s1 += x1 * x1;
    }
    atomicAdd(&nrm[c0], s0);
    atomicAdd(&nrm[c1], s1);
}

// ---------------- normalize + cl1 (relu) + maxpool(8) -> pooled (B, 32768) ----------------
__global__ void gcn_head_kernel(const float* __restrict__ OutAcc, const float* __restrict__ nrm,
                                const float* __restrict__ w, const float* __restrict__ bias,
                                float* __restrict__ pooled) {
    __shared__ float wsh[CL1F * F0];
    __shared__ float invn[F0];
    int blk = blockIdx.x;                 // 4096 blocks: 32 b * 128 vp-groups
    int b  = blk >> 7;
    int vp = ((blk & 127) << 3) + (threadIdx.x >> 5);   // 0..1023
    int o  = threadIdx.x & 31;
    if (threadIdx.x < CL1F * F0) wsh[threadIdx.x] = w[threadIdx.x];
    if (threadIdx.x < F0) {
        float s = nrm[b * F0 + threadIdx.x];
        invn[threadIdx.x] = 1.0f / fmaxf(sqrtf(s), 1e-12f);
    }
    __syncthreads();
    float bo = bias[o];
    float best = 0.f;                     // relu outputs are >= 0
    for (int pv = 0; pv < POOLW; ++pv) {
        int v = vp * POOLW + pv;
        const float* row = OutAcc + (size_t)v * NCOL + b * F0;
        float acc = bo;
#pragma unroll
        for (int f = 0; f < F0; ++f)
            acc += row[f] * invn[f] * wsh[o * F0 + f];
        best = fmaxf(best, fmaxf(acc, 0.f));
    }
    pooled[(size_t)b * FC1FIN + (size_t)vp * CL1F + o] = best;
}

// ---------------- memory-bound FC: one block per output feature, 32 batch accumulators ----
__global__ __launch_bounds__(256) void fc_bigK_kernel(
    const float* __restrict__ X,    // (BB, Kdim)
    const float* __restrict__ Wt,   // (F, Kdim) — each row read exactly once
    const float* __restrict__ bias,
    float* __restrict__ Y,          // (BB, F), relu applied
    int Kdim, int F)
{
    int o = blockIdx.x;
    const float* wr = Wt + (size_t)o * Kdim;
    float acc[BB];
#pragma unroll
    for (int b = 0; b < BB; ++b) acc[b] = 0.f;
    for (int k = threadIdx.x; k < Kdim; k += 256) {
        __builtin_prefetch(wr + k + 4096, 0, 1);   // global_prefetch_b8 on the weight stream
        float wv = wr[k];
#pragma unroll
        for (int b = 0; b < BB; ++b)
            acc[b] += wv * X[(size_t)b * Kdim + k];
    }
    __shared__ float red[256];
    for (int b = 0; b < BB; ++b) {
        red[threadIdx.x] = acc[b];
        __syncthreads();
        for (int s = 128; s > 0; s >>= 1) {
            if (threadIdx.x < s) red[threadIdx.x] += red[threadIdx.x + s];
            __syncthreads();
        }
        if (threadIdx.x == 0) Y[(size_t)b * F + o] = fmaxf(red[0] + bias[o], 0.f);
        __syncthreads();
    }
}

// ---------------- train-mode BatchNorm1d over batch (biased stats), in place ----------------
__global__ void bn_kernel(float* __restrict__ h, const float* __restrict__ g,
                          const float* __restrict__ be, int Bn, int F) {
    int f = blockIdx.x * blockDim.x + threadIdx.x;
    if (f >= F) return;
    float m = 0.f;
    for (int b = 0; b < Bn; ++b) m += h[(size_t)b * F + f];
    m /= (float)Bn;
    float v = 0.f;
    for (int b = 0; b < Bn; ++b) { float d = h[(size_t)b * F + f] - m; v += d * d; }
    v /= (float)Bn;
    float inv = rsqrtf(v + BN_EPS);
    for (int b = 0; b < Bn; ++b)
        h[(size_t)b * F + f] = (h[(size_t)b * F + f] - m) * inv * g[f] + be[f];
}

// ---------------- small FC (relu optional) ----------------
__global__ void fc_small_kernel(const float* __restrict__ X, const float* __restrict__ Wt,
                                const float* __restrict__ bias, float* __restrict__ Y,
                                int Kdim, int F, int relu) {
    int t = blockIdx.x * blockDim.x + threadIdx.x;
    if (t >= BB * F) return;
    int b = t / F, o = t % F;
    float acc = bias[o];
    for (int k = 0; k < Kdim; ++k)
        acc += X[(size_t)b * Kdim + k] * Wt[(size_t)o * Kdim + k];
    Y[t] = relu ? fmaxf(acc, 0.f) : acc;
}

// ---------------- fusion head: logits = [x_hidden, h2] @ sum2_w^T + b ----------------
__global__ void head_kernel(const float* __restrict__ xh, const float* __restrict__ h2,
                            const float* __restrict__ Wt, const float* __restrict__ bias,
                            float* __restrict__ out) {
    int t = blockIdx.x * blockDim.x + threadIdx.x;
    if (t >= BB * OUTF) return;
    int b = t / OUTF, o = t % OUTF;
    const float* wr = Wt + (size_t)o * (FC1F + NNF2);
    float acc = bias[o];
    for (int k = 0; k < FC1F; ++k) acc += xh[(size_t)b * FC1F + k] * wr[k];
    for (int k = 0; k < NNF2; ++k) acc += h2[(size_t)b * NNF2 + k] * wr[FC1F + k];
    out[t] = acc;
}

// ==================================================================================
extern "C" void kernel_launch(void* const* d_in, const int* in_sizes, int n_in,
                              void* d_out, int out_size, void* d_ws, size_t ws_size,
                              hipStream_t stream) {
    (void)in_sizes; (void)n_in; (void)out_size; (void)ws_size;
    const float* x_in  = (const float*)d_in[0];
    const float* L     = (const float*)d_in[1];
    const float* theta = (const float*)d_in[2];
    const float* cl1_w = (const float*)d_in[3];
    const float* cl1_b = (const float*)d_in[4];
    const float* fc1_w = (const float*)d_in[5];
    const float* fc1_b = (const float*)d_in[6];
    const float* nn1_w = (const float*)d_in[7];
    const float* nn1_b = (const float*)d_in[8];
    const float* g1    = (const float*)d_in[9];
    const float* be1   = (const float*)d_in[10];
    const float* nn2_w = (const float*)d_in[11];
    const float* nn2_b = (const float*)d_in[12];
    const float* g2    = (const float*)d_in[13];
    const float* be2   = (const float*)d_in[14];
    const float* s2_w  = (const float*)d_in[15];
    const float* s2_b  = (const float*)d_in[16];
    float* out = (float*)d_out;

    // workspace layout (~168 MB total)
    char* ws = (char*)d_ws;
    float*    a      = (float*)(ws + 0);                 // 9 f32
    float*    nrm    = (float*)(ws + (1  << 10));        // 512 f32
    float*    h1     = (float*)(ws + (4  << 10));        // 32x256
    float*    h2     = (float*)(ws + (40 << 10));        // 32x128
    float*    xh     = (float*)(ws + (60 << 10));        // 32x512
    float*    pooled = (float*)(ws + (128 << 10));       // 32x32768 (4 MB)
    float*    OutAcc = (float*)(ws + (8ull  << 20));     // V x 512 f32 (16 MB)
    uint16_t* W0t    = (uint16_t*)(ws + (24ull << 20));  // 512 x V bf16 (8 MB)
    uint16_t* W1t    = (uint16_t*)(ws + (32ull << 20));  // 512 x V bf16 (8 MB)
    uint16_t* Lb     = (uint16_t*)(ws + (40ull << 20));  // V x V bf16 (128 MB, L2-resident)

    bern_kernel<<<1, 32, 0, stream>>>(theta, a);

    long long n8 = (long long)V * V / 8;
    cvt_bf16_kernel<<<(int)(n8 / 256), 256, 0, stream>>>(L, Lb, n8);
    init_acc_kernel<<<(V * NCOL) / 256, 256, 0, stream>>>(x_in, a, OutAcc);
    init_wt_kernel<<<(NCOL * V) / 256, 256, 0, stream>>>(x_in, W0t);

    uint16_t* Wc = W0t; uint16_t* Wn = W1t;
    for (int j = 1; j <= KPOLY; ++j) {
        gemm_bf16_kernel<<<dim3(V / BM, NCOL / BN), 256, 0, stream>>>(Lb, Wc, Wn, OutAcc, a, j);
        uint16_t* t = Wc; Wc = Wn; Wn = t;
    }

    zero_kernel<<<2, 256, 0, stream>>>(nrm, NCOL);
    colsumsq_kernel<<<V / 128, 256, 0, stream>>>(OutAcc, nrm);
    gcn_head_kernel<<<(BB * (V / POOLW)) / 8, 256, 0, stream>>>(OutAcc, nrm, cl1_w, cl1_b, pooled);

    fc_bigK_kernel<<<FC1F, 256, 0, stream>>>(pooled, fc1_w, fc1_b, xh, FC1FIN, FC1F);
    fc_bigK_kernel<<<NNF1, 256, 0, stream>>>(x_in, nn1_w, nn1_b, h1, V * F0, NNF1);
    bn_kernel<<<1, NNF1, 0, stream>>>(h1, g1, be1, BB, NNF1);
    fc_small_kernel<<<(BB * NNF2 + 255) / 256, 256, 0, stream>>>(h1, nn2_w, nn2_b, h2, NNF1, NNF2, 1);
    bn_kernel<<<1, NNF2, 0, stream>>>(h2, g2, be2, BB, NNF2);
    head_kernel<<<(BB * OUTF + 255) / 256, 256, 0, stream>>>(xh, h2, s2_w, s2_b, out);
}
